// Linear8bit_37950331027660
// MI455X (gfx1250) — 8-bit matrix operand order repaired and checked by reading, compiled, not run on hardware
//
#include <hip/hip_runtime.h>
#include <cstdint>

typedef __attribute__((ext_vector_type(8))) int v8i;
typedef int v4i_vs __attribute__((vector_size(16)));

#define TILE_M 128
#define TILE_N 128
#define TILE_K 64
#define LDSS   80

#if defined(__has_builtin)
#if __has_builtin(__builtin_amdgcn_global_load_async_to_lds_b128) && \
    __has_builtin(__builtin_amdgcn_s_wait_asynccnt)
#define USE_ASYNC_LDS 1
#endif
#endif
#ifndef USE_ASYNC_LDS
#define USE_ASYNC_LDS 0
#endif

#if USE_ASYNC_LDS
typedef __attribute__((address_space(1))) v4i_vs* gas_ptr;
typedef __attribute__((address_space(3))) v4i_vs* lds_ptr;
static __device__ __forceinline__ gas_ptr to_gas(const void* p) {
    return (gas_ptr)(unsigned long long)(uintptr_t)p;
}
static __device__ __forceinline__ lds_ptr to_lds(void* p) {
    return (lds_ptr)(unsigned int)(uintptr_t)p;
}
#endif

__global__ __launch_bounds__(256) void quant_rows_kernel(
    const float* __restrict__ in, int8_t* __restrict__ q,
    float* __restrict__ dq, int K)
{
    const int row = blockIdx.x;
    const int tid = threadIdx.x;
    const float* r = in + (size_t)row * K;

    float amax = 0.0f;
    for (int i = tid * 4; i < K; i += 256 * 4) {
        float4 v = *(const float4*)(r + i);
        amax = fmaxf(amax, fmaxf(fmaxf(fabsf(v.x), fabsf(v.y)),
                                 fmaxf(fabsf(v.z), fabsf(v.w))));
    }

    __shared__ float red[256];
    red[tid] = amax;
    __syncthreads();
    #pragma unroll
    for (int s = 128; s > 0; s >>= 1) {
        if (tid < s) red[tid] = fmaxf(red[tid], red[tid + s]);
        __syncthreads();
    }
    const float absmax = fmaxf(red[0], 1e-8f);
    const float scale  = 127.0f / absmax;
    if (tid == 0) dq[row] = absmax / 127.0f;

    int8_t* qr = q + (size_t)row * K;
    for (int i = tid * 4; i < K; i += 256 * 4) {
        float4 v = *(const float4*)(r + i);
        int a = (int)rintf(fminf(fmaxf(v.x * scale, -127.0f), 127.0f));
        int b = (int)rintf(fminf(fmaxf(v.y * scale, -127.0f), 127.0f));
        int c = (int)rintf(fminf(fmaxf(v.z * scale, -127.0f), 127.0f));
        int d = (int)rintf(fminf(fmaxf(v.w * scale, -127.0f), 127.0f));
        int packed = (a & 0xff) | ((b & 0xff) << 8) | ((c & 0xff) << 16) | ((d & 0xff) << 24);
        *(int*)(qr + i) = packed;
    }
}

__global__ __launch_bounds__(256, 2) void gemm_i8_wmma_kernel(
    const int8_t* __restrict__ Aq, const int8_t* __restrict__ Bq,
    const float* __restrict__ sa, const float* __restrict__ sb,
    const float* __restrict__ bias, float* __restrict__ out,
    int M, int N, int K)
{
    __shared__ int8_t lA[2][TILE_M * LDSS];
    __shared__ int8_t lB[2][TILE_N * LDSS];

    const int tid  = threadIdx.x;
    const int lane = tid & 31;
    const int wave = tid >> 5;
    const int wm   = wave & 1;
    const int wn   = wave >> 1;

    const int blockN = blockIdx.x * TILE_N;
    const int blockM = blockIdx.y * TILE_M;

    v8i acc[4][2];
    #pragma unroll
    for (int mt = 0; mt < 4; ++mt)
        #pragma unroll
        for (int nt = 0; nt < 2; ++nt)
            #pragma unroll
            for (int i = 0; i < 8; ++i)
                acc[mt][nt][i] = 0;

    const int ldRow = tid >> 2;
    const int ldCol = (tid & 3) * 16;
    const size_t gA0 = (size_t)(blockM + ldRow)      * K + ldCol;
    const size_t gA1 = (size_t)(blockM + ldRow + 64) * K + ldCol;
    const size_t gB0 = (size_t)(blockN + ldRow)      * K + ldCol;
    const size_t gB1 = (size_t)(blockN + ldRow + 64) * K + ldCol;
    const int    lOff0 = ldRow * LDSS + ldCol;
    const int    lOff1 = (ldRow + 64) * LDSS + ldCol;

    const int aRow   = lane & 15;
    const int aKbase = (lane >> 4) * 8;
    const int bCol   = lane & 15;
    const int bKhalf = (lane >> 4) * 16;

#if USE_ASYNC_LDS
    __builtin_amdgcn_global_load_async_to_lds_b128(to_gas(Aq + gA0), to_lds(&lA[0][lOff0]), 0, 0);
    __builtin_amdgcn_global_load_async_to_lds_b128(to_gas(Aq + gA1), to_lds(&lA[0][lOff1]), 0, 0);
    __builtin_amdgcn_global_load_async_to_lds_b128(to_gas(Bq + gB0), to_lds(&lB[0][lOff0]), 0, 0);
    __builtin_amdgcn_global_load_async_to_lds_b128(to_gas(Bq + gB1), to_lds(&lB[0][lOff1]), 0, 0);
    __builtin_amdgcn_s_wait_asynccnt(0);
#else
    {
        int4 a0 = *(const int4*)(Aq + gA0);
        int4 a1 = *(const int4*)(Aq + gA1);
        int4 b0 = *(const int4*)(Bq + gB0);
        int4 b1 = *(const int4*)(Bq + gB1);
        *(int4*)(&lA[0][lOff0]) = a0;
        *(int4*)(&lA[0][lOff1]) = a1;
        *(int4*)(&lB[0][lOff0]) = b0;
        *(int4*)(&lB[0][lOff1]) = b1;
    }
#endif

    int buf = 0;
    for (int k0 = 0; k0 < K; k0 += TILE_K) {
        __syncthreads();
        const bool has_next = (k0 + TILE_K) < K;
        const int  kn = k0 + TILE_K;

#if USE_ASYNC_LDS
        if (has_next) {
            __builtin_amdgcn_global_load_async_to_lds_b128(to_gas(Aq + gA0 + kn), to_lds(&lA[buf ^ 1][lOff0]), 0, 0);
            __builtin_amdgcn_global_load_async_to_lds_b128(to_gas(Aq + gA1 + kn), to_lds(&lA[buf ^ 1][lOff1]), 0, 0);
            __builtin_amdgcn_global_load_async_to_lds_b128(to_gas(Bq + gB0 + kn), to_lds(&lB[buf ^ 1][lOff0]), 0, 0);
            __builtin_amdgcn_global_load_async_to_lds_b128(to_gas(Bq + gB1 + kn), to_lds(&lB[buf ^ 1][lOff1]), 0, 0);
        }
#else
        int4 a0, a1, b0, b1;
        if (has_next) {
            a0 = *(const int4*)(Aq + gA0 + kn);
            a1 = *(const int4*)(Aq + gA1 + kn);
            b0 = *(const int4*)(Bq + gB0 + kn);
            b1 = *(const int4*)(Bq + gB1 + kn);
        }
#endif

        v8i afrag[4];
        #pragma unroll
        for (int mt = 0; mt < 4; ++mt) {
            const int8_t* p = &lA[buf][(wm * 64 + mt * 16 + aRow) * LDSS];
            #pragma unroll
            for (int i = 0; i < 8; ++i) {
                const int k = aKbase + (i >> 1) * 16 + (i & 1) * 4;
                afrag[mt][i] = *(const int*)(p + k);
            }
        }
        v8i bfrag[2];
        #pragma unroll
        for (int nt = 0; nt < 2; ++nt) {
            const int8_t* p = &lB[buf][(wn * 32 + nt * 16 + bCol) * LDSS];
            #pragma unroll
            for (int i = 0; i < 8; ++i) {
                const int k = (i >> 1) * 16 + (bKhalf >> 1) + (i & 1) * 4;
                bfrag[nt][i] = *(const int*)(p + k);
            }
        }

        #pragma unroll
        for (int mt = 0; mt < 4; ++mt)
            #pragma unroll
            for (int nt = 0; nt < 2; ++nt)
                acc[mt][nt] = __builtin_amdgcn_wmma_i32_16x16x64_iu8(
                     true, afrag[mt],
                     true, bfrag[nt],
                    acc[mt][nt],  false,  false);

        if (has_next) {
#if USE_ASYNC_LDS
            __builtin_amdgcn_s_wait_asynccnt(0);
#else
            *(int4*)(&lA[buf ^ 1][lOff0]) = a0;
            *(int4*)(&lA[buf ^ 1][lOff1]) = a1;
            *(int4*)(&lB[buf ^ 1][lOff0]) = b0;
            *(int4*)(&lB[buf ^ 1][lOff1]) = b1;
#endif
            buf ^= 1;
        }
    }

    const int nlocal = lane & 15;
    const int mhalf  = (lane >> 4) * 8;
    #pragma unroll
    for (int mt = 0; mt < 4; ++mt) {
        float sav[8];
        #pragma unroll
        for (int v = 0; v < 8; ++v)
            sav[v] = sa[blockM + wm * 64 + mt * 16 + mhalf + v];
        #pragma unroll
        for (int nt = 0; nt < 2; ++nt) {
            const int gn  = blockN + wn * 32 + nt * 16 + nlocal;
            const float sbn = sb[gn];
            const float bn  = bias[gn];
            #pragma unroll
            for (int v = 0; v < 8; ++v) {
                const int gm = blockM + wm * 64 + mt * 16 + mhalf + v;
                out[(size_t)gm * N + gn] =
                    (float)acc[mt][nt][v] * sav[v] * sbn + bn;
            }
        }
    }
}

extern "C" void kernel_launch(void* const* d_in, const int* in_sizes, int n_in,
                              void* d_out, int out_size, void* d_ws, size_t ws_size,
                              hipStream_t stream) {
    const float* x      = (const float*)d_in[0];
    const float* weight = (const float*)d_in[1];
    const float* bias   = (const float*)d_in[2];
    float* out          = (float*)d_out;

    const int N = in_sizes[2];
    const int K = in_sizes[1] / N;
    const int M = in_sizes[0] / K;

    int8_t* xq = (int8_t*)d_ws;
    int8_t* wq = xq + (size_t)M * K;
    float*  sa = (float*)(wq + (size_t)N * K);
    float*  sb = sa + M;

    quant_rows_kernel<<<M, 256, 0, stream>>>(x,      xq, sa, K);
    quant_rows_kernel<<<N, 256, 0, stream>>>(weight, wq, sb, K);

    dim3 grid(N / TILE_N, M / TILE_M);
    gemm_i8_wmma_kernel<<<grid, 256, 0, stream>>>(xq, wq, sa, sb, bias, out,
                                                  M, N, K);
}
